// EGNNVectorField_2010044695217
// MI455X (gfx1250) — compile-verified, matrix operand order checked by reading, not run on hardware
//
#include <hip/hip_runtime.h>
#include <hip/hip_bf16.h>

typedef __bf16 bf16;
typedef bf16 v16bf __attribute__((ext_vector_type(16)));
typedef bf16 v8bf  __attribute__((ext_vector_type(8)));
typedef float v8f  __attribute__((ext_vector_type(8)));
typedef unsigned short u16t;

#define N_PTS   16384
#define NNODES  20480
#define GRID_G  512
#define CD      128
#define DEG     16
#define WAVES   4
#define PI_F    3.14159265358979f

__device__ __forceinline__ u16t f2bf(float f) {
  union { float f; unsigned u; } v; v.f = f;
  unsigned r = v.u + 0x7FFFu + ((v.u >> 16) & 1u);   // round-to-nearest-even
  return (u16t)(r >> 16);
}
__device__ __forceinline__ float silu_f(float x) { return x / (1.0f + __expf(-x)); }

// Wave-level GEMM: C(16 x 128) += A(16 x 32*kchunks) @ W, A rows contiguous bf16 in LDS,
// W pre-packed in global memory in per-lane B-fragment order (32B contiguous per lane/tile).
__device__ __forceinline__ void wave_gemm(const u16t* ldsA, int strideA,
                                          const u16t* wpk, int kchunks,
                                          v8f acc[8], int lane) {
  int m  = lane & 15;
  int hi = (lane >> 4) & 1;
  for (int kc = 0; kc < kchunks; ++kc) {
    const u16t* arow = ldsA + m * strideA + kc * 32 + hi * 8;
    v8bf alo = *(const v8bf*)(arow);        // K = kbase+0..7
    v8bf ahi = *(const v8bf*)(arow + 16);   // K = kbase+16..23
    v16bf a = __builtin_shufflevector(alo, ahi, 0,1,2,3,4,5,6,7,8,9,10,11,12,13,14,15);
    const u16t* bp = wpk + (kc * 8) * 512 + lane * 16;
#pragma unroll
    for (int nt = 0; nt < 8; ++nt) {
      v16bf b = *(const v16bf*)(bp + nt * 512);
      acc[nt] = __builtin_amdgcn_wmma_f32_16x16x32_bf16(
          false, a, false, b, (short)0, acc[nt], false, false);
    }
  }
}

// ---- init: x = [qp ; grid broadcast], h = [0 ; codes], m_aggr = 0 --------------------
__global__ void init_kernel(const float* __restrict__ qp, const float* __restrict__ codes,
                            const float* __restrict__ gp,
                            float* __restrict__ x, float* __restrict__ h,
                            float* __restrict__ magg) {
  int n = blockIdx.x * blockDim.x + threadIdx.x;
  if (n >= NNODES) return;
  if (n < N_PTS) {
    x[n*3+0] = qp[n*3+0]; x[n*3+1] = qp[n*3+1]; x[n*3+2] = qp[n*3+2];
    for (int c = 0; c < CD; ++c) { h[n*CD+c] = 0.0f; magg[n*CD+c] = 0.0f; }
  } else {
    int t = n - N_PTS; int g = t % GRID_G;
    x[n*3+0] = gp[g*3+0]; x[n*3+1] = gp[g*3+1]; x[n*3+2] = gp[g*3+2];
    for (int c = 0; c < CD; ++c) { h[n*CD+c] = codes[t*CD+c]; magg[n*CD+c] = 0.0f; }
  }
}

// ---- pack: fp32 row-major (K x 128) weight -> bf16 WMMA B-fragment layout -----------
__global__ void pack_kernel(const float* __restrict__ W, u16t* __restrict__ out, int ktiles) {
  int tile = blockIdx.x;                    // ktiles*8 tiles
  int kt = tile >> 3, nt = tile & 7;
  int lane = threadIdx.x & 31;
  int n  = nt * 16 + (lane & 15);
  int kb = kt * 32 + ((lane >> 4) & 1) * 8;
  u16t* dst = out + (tile * 32 + lane) * 16;
#pragma unroll
  for (int e = 0; e < 16; ++e) {
    int k = kb + ((e < 8) ? 0 : 16) + (e & 7);
    dst[e] = f2bf(W[k * 128 + n]);
  }
  (void)ktiles;
}

// ---- edge layer: one wave per query point (16 edges = one WMMA M-tile) --------------
template <int TOUT>
__launch_bounds__(128)
__global__ void edge_kernel(float* __restrict__ x, const float* __restrict__ h,
                            float* __restrict__ magg, const int* __restrict__ erow,
                            const u16t* __restrict__ e1p, const float* __restrict__ eW1l,
                            const float* __restrict__ eb1l,
                            const u16t* __restrict__ e2p, const float* __restrict__ eb2l,
                            const float* __restrict__ coefW, const float* __restrict__ coefB,
                            float* __restrict__ out, const float* __restrict__ qp0) {
  __shared__ __align__(16) u16t s_feat[WAVES][16*264]; // [edge][256 feats], stride-padded
  __shared__ __align__(16) u16t s_m1[WAVES][16*136];   // hidden1 bf16
  __shared__ float s_m[WAVES][16*132];                 // final message f32
  __shared__ float s_dist[WAVES][16];
  __shared__ float s_cm[WAVES][16];
  __shared__ float s_dir[WAVES][16][3];
  __shared__ int   s_row[WAVES][16];
  __shared__ float s_red[WAVES][16][16];

  const int wave = threadIdx.x >> 5, lane = threadIdx.x & 31;
  const int q = blockIdx.x * WAVES + wave;
  const int n0 = lane & 15, hi = (lane >> 4) & 1;

  // geometry: lanes 0..15 each own one edge
  if (lane < 16) {
    int r = erow[q * DEG + lane];
    s_row[wave][lane] = r;
    float rx = x[r*3+0] - x[q*3+0];
    float ry = x[r*3+1] - x[q*3+1];
    float rz = x[r*3+2] - x[q*3+2];
    float d = sqrtf(rx*rx + ry*ry + rz*rz);
    s_dist[wave][lane] = d;
    float inv = 1.0f / (d + 1e-8f);
    s_dir[wave][lane][0] = rx * inv;
    s_dir[wave][lane][1] = ry * inv;
    s_dir[wave][lane][2] = rz * inv;
    s_cm[wave][lane] = (d <= 10.0f) ? 0.5f * (__cosf(d * (PI_F / 10.0f)) + 1.0f) : 0.0f;
  }
  __syncthreads();

  // gather edge features: [h[row](128) | h[col](128)] as bf16 rows
  const float4* h4 = (const float4*)h;
  float4 hq = h4[q * 32 + lane];
  u16t hqb0 = f2bf(hq.x), hqb1 = f2bf(hq.y), hqb2 = f2bf(hq.z), hqb3 = f2bf(hq.w);
  for (int j = 0; j < 16; ++j) {
    int r = s_row[wave][j];
    float4 hv = h4[r * 32 + lane];
    u16t* d0 = &s_feat[wave][j * 264 + lane * 4];
    d0[0] = f2bf(hv.x); d0[1] = f2bf(hv.y); d0[2] = f2bf(hv.z); d0[3] = f2bf(hv.w);
    u16t* d1 = &s_feat[wave][j * 264 + 128 + lane * 4];
    d1[0] = hqb0; d1[1] = hqb1; d1[2] = hqb2; d1[3] = hqb3;
  }
  __syncthreads();

  // GEMM1: bias + dist * W1[256,:] (rank-1) + A@W1[0:256,:]
  float distm[8];
#pragma unroll
  for (int v = 0; v < 8; ++v) distm[v] = s_dist[wave][hi*8 + v];
  v8f acc[8];
  const float* wd = eW1l + 256 * 128;
#pragma unroll
  for (int nt = 0; nt < 8; ++nt) {
    float b = eb1l[nt*16 + n0], w = wd[nt*16 + n0];
#pragma unroll
    for (int v = 0; v < 8; ++v) acc[nt][v] = b + distm[v] * w;
  }
  wave_gemm(&s_feat[wave][0], 264, e1p, 8, acc, lane);
#pragma unroll
  for (int nt = 0; nt < 8; ++nt)
#pragma unroll
    for (int v = 0; v < 8; ++v)
      s_m1[wave][(hi*8 + v) * 136 + nt*16 + n0] = f2bf(silu_f(acc[nt][v]));
  __syncthreads();

  // GEMM2 + silu + cosine cutoff
  v8f acc2[8];
#pragma unroll
  for (int nt = 0; nt < 8; ++nt) {
    float b = eb2l[nt*16 + n0];
#pragma unroll
    for (int v = 0; v < 8; ++v) acc2[nt][v] = b;
  }
  wave_gemm(&s_m1[wave][0], 136, e2p, 4, acc2, lane);
  float cmm[8];
#pragma unroll
  for (int v = 0; v < 8; ++v) cmm[v] = s_cm[wave][hi*8 + v];
#pragma unroll
  for (int nt = 0; nt < 8; ++nt)
#pragma unroll
    for (int v = 0; v < 8; ++v)
      s_m[wave][(hi*8 + v) * 132 + nt*16 + n0] = silu_f(acc2[nt][v]) * cmm[v];
  __syncthreads();

  if (TOUT == 1) { // message aggregation (only needed when h still updates)
#pragma unroll
    for (int c0 = 0; c0 < 4; ++c0) {
      int c = lane * 4 + c0; float s = 0.0f;
      for (int j = 0; j < 16; ++j) s += s_m[wave][j * 132 + c];
      magg[q * CD + c] = s * (1.0f / 16.0f);
    }
  }

  // coef projection + directional messages
  if (lane < 16) {
    int j = lane;
    float cf[TOUT];
#pragma unroll
    for (int t = 0; t < TOUT; ++t) cf[t] = coefB[t];
    for (int k = 0; k < 128; ++k) {
      float mv = s_m[wave][j * 132 + k];
#pragma unroll
      for (int t = 0; t < TOUT; ++t) cf[t] += mv * coefW[k * TOUT + t];
    }
#pragma unroll
    for (int t = 0; t < TOUT; ++t)
#pragma unroll
      for (int d = 0; d < 3; ++d)
        s_red[wave][j][t*3 + d] = cf[t] * s_dir[wave][j][d];
  }
  __syncthreads();

  if (TOUT == 1) {
    if (lane < 3) {
      float s = 0.0f;
      for (int j = 0; j < 16; ++j) s += s_red[wave][j][lane];
      x[q*3 + lane] += s * (1.0f / 16.0f);
    }
  } else {
    if (lane < 3 * TOUT) {
      int t = lane / 3, d = lane % 3;
      float s = 0.0f;
      for (int j = 0; j < 16; ++j) s += s_red[wave][j][lane];
      out[(q * TOUT + t) * 3 + d] = x[q*3 + d] + s * (1.0f / 16.0f) - qp0[q*3 + d];
    }
  }
}

// ---- node layer: one wave per 16-node tile ------------------------------------------
__launch_bounds__(128)
__global__ void node_kernel(float* __restrict__ h, const float* __restrict__ magg,
                            const u16t* __restrict__ n1p, const float* __restrict__ bn1l,
                            const u16t* __restrict__ n2p, const float* __restrict__ bn2l) {
  __shared__ __align__(16) u16t s_in[WAVES][16*264];
  __shared__ __align__(16) u16t s_t1[WAVES][16*136];
  const int wave = threadIdx.x >> 5, lane = threadIdx.x & 31;
  const int nb = (blockIdx.x * WAVES + wave) * 16;
  const int n0 = lane & 15, hi = (lane >> 4) & 1;
  const float4* h4 = (const float4*)h;
  const float4* m4 = (const float4*)magg;
  for (int j = 0; j < 16; ++j) {
    int nd = nb + j;
    float4 hv = h4[nd * 32 + lane];
    u16t* d0 = &s_in[wave][j * 264 + lane * 4];
    d0[0] = f2bf(hv.x); d0[1] = f2bf(hv.y); d0[2] = f2bf(hv.z); d0[3] = f2bf(hv.w);
    float4 mv = m4[nd * 32 + lane];
    u16t* d1 = &s_in[wave][j * 264 + 128 + lane * 4];
    d1[0] = f2bf(mv.x); d1[1] = f2bf(mv.y); d1[2] = f2bf(mv.z); d1[3] = f2bf(mv.w);
  }
  __syncthreads();
  v8f acc[8];
#pragma unroll
  for (int nt = 0; nt < 8; ++nt) {
    float b = bn1l[nt*16 + n0];
#pragma unroll
    for (int v = 0; v < 8; ++v) acc[nt][v] = b;
  }
  wave_gemm(&s_in[wave][0], 264, n1p, 8, acc, lane);
#pragma unroll
  for (int nt = 0; nt < 8; ++nt)
#pragma unroll
    for (int v = 0; v < 8; ++v)
      s_t1[wave][(hi*8 + v) * 136 + nt*16 + n0] = f2bf(silu_f(acc[nt][v]));
  __syncthreads();
  v8f acc2[8];
#pragma unroll
  for (int nt = 0; nt < 8; ++nt) {
    float b = bn2l[nt*16 + n0];
#pragma unroll
    for (int v = 0; v < 8; ++v) acc2[nt][v] = b;
  }
  wave_gemm(&s_t1[wave][0], 136, n2p, 4, acc2, lane);
#pragma unroll
  for (int nt = 0; nt < 8; ++nt)
#pragma unroll
    for (int v = 0; v < 8; ++v) {
      int idx = (nb + hi*8 + v) * CD + nt*16 + n0;
      h[idx] += acc2[nt][v];
    }
}

// ---- host orchestration --------------------------------------------------------------
extern "C" void kernel_launch(void* const* d_in, const int* in_sizes, int n_in,
                              void* d_out, int out_size, void* d_ws, size_t ws_size,
                              hipStream_t stream) {
  (void)in_sizes; (void)n_in; (void)out_size; (void)ws_size;
  const float* qp    = (const float*)d_in[0];
  const float* codes = (const float*)d_in[1];
  const float* gp    = (const float*)d_in[2];
  const float* eW1   = (const float*)d_in[3];
  const float* eb1   = (const float*)d_in[4];
  const float* eW2   = (const float*)d_in[5];
  const float* eb2   = (const float*)d_in[6];
  const float* cW    = (const float*)d_in[7];
  const float* cb    = (const float*)d_in[8];
  const float* fcW   = (const float*)d_in[9];
  const float* fcb   = (const float*)d_in[10];
  const float* nW1   = (const float*)d_in[11];
  const float* bn1   = (const float*)d_in[12];
  const float* nW2   = (const float*)d_in[13];
  const float* bn2   = (const float*)d_in[14];
  const int*   erow  = (const int*)d_in[15]; // edge_index[0] = first E entries
  float* out = (float*)d_out;

  char* ws = (char*)d_ws;
  float* x    = (float*)(ws);                              //  245,760 B
  float* h    = (float*)(ws + 245760);                     // 10,485,760 B
  float* magg = (float*)(ws + 245760 + 10485760);          // 10,485,760 B
  u16t*  wp   = (u16t*)(ws + 245760 + 2u*10485760u);       //   786,432 B packed weights

  init_kernel<<<(NNODES + 127) / 128, 128, 0, stream>>>(qp, codes, gp, x, h, magg);

  for (int i = 0; i < 4; ++i) {
    u16t* e1p = wp + (size_t)i * 98304;
    u16t* e2p = e1p + 32768;
    u16t* n1p = e2p + 16384;
    u16t* n2p = n1p + 32768;
    pack_kernel<<<64, 32, 0, stream>>>(eW1 + (size_t)i * 257 * 128, e1p, 8);
    pack_kernel<<<32, 32, 0, stream>>>(eW2 + (size_t)i * 128 * 128, e2p, 4);
    pack_kernel<<<64, 32, 0, stream>>>(nW1 + (size_t)i * 256 * 128, n1p, 8);
    pack_kernel<<<32, 32, 0, stream>>>(nW2 + (size_t)i * 128 * 128, n2p, 4);
  }

  for (int i = 0; i < 3; ++i) {
    u16t* e1p = wp + (size_t)i * 98304;
    u16t* e2p = e1p + 32768;
    u16t* n1p = e2p + 16384;
    u16t* n2p = n1p + 32768;
    edge_kernel<1><<<N_PTS / WAVES, 128, 0, stream>>>(
        x, h, magg, erow,
        e1p, eW1 + (size_t)i * 257 * 128, eb1 + i * 128,
        e2p, eb2 + i * 128,
        cW + i * 128, cb + i, nullptr, qp);
    node_kernel<<<NNODES / 16 / WAVES, 128, 0, stream>>>(
        h, magg, n1p, bn1 + i * 128, n2p, bn2 + i * 128);
  }

  { // final layer: T=5 coef columns, writes output directly
    u16t* e1p = wp + (size_t)3 * 98304;
    u16t* e2p = e1p + 32768;
    edge_kernel<5><<<N_PTS / WAVES, 128, 0, stream>>>(
        x, h, magg, erow,
        e1p, eW1 + (size_t)3 * 257 * 128, eb1 + 3 * 128,
        e2p, eb2 + 3 * 128,
        fcW, fcb, out, qp);
  }
}
